// GIN_89919435309328
// MI455X (gfx1250) — matrix operand order repaired and checked by reading, not compiled, not run on hardware
//
#include <hip/hip_runtime.h>
#include <hip/hip_fp16.h>
#include <math.h>

#define BN_EPS 1e-5f
#define FDIM 64     // input features
#define HDIM 64     // hidden
#define H2   128    // 2*H
#define CDIM 40     // classes
#define CPAD 48     // classes padded to multiple of 16
#define WAVES 4     // wave32s per block

typedef __attribute__((ext_vector_type(16))) _Float16 v16h;
typedef __attribute__((ext_vector_type(8)))  _Float16 v8h;
typedef __attribute__((ext_vector_type(8)))  float    v8f;
typedef __attribute__((ext_vector_type(4)))  int      v4i;

typedef __attribute__((address_space(1))) v4i gv4i;   // global v4i
typedef __attribute__((address_space(3))) v4i lv4i;   // LDS v4i

// ---- CDNA5 async Global<->LDS (guarded; fallback = plain vector copies) ----
#if defined(__gfx1250__) && __has_builtin(__builtin_amdgcn_global_load_async_to_lds_b128)
#define HAVE_ASYNC_LOAD 1
#endif
#if defined(__gfx1250__) && __has_builtin(__builtin_amdgcn_global_store_async_from_lds_b128)
#define HAVE_ASYNC_STORE 1
#endif

__device__ __forceinline__ void wait_async0() {
#if defined(__gfx1250__)
#if __has_builtin(__builtin_amdgcn_s_wait_asynccnt)
  __builtin_amdgcn_s_wait_asynccnt(0);
  asm volatile("" ::: "memory");
#else
  asm volatile("s_wait_asynccnt 0x0" ::: "memory");
#endif
#endif
}

// ---------------------------------------------------------------------------
// x (f32) -> xh (f16)
// ---------------------------------------------------------------------------
__global__ void gin_cvt_h(const float* __restrict__ x, _Float16* __restrict__ xh, long n) {
  long i = (long)blockIdx.x * blockDim.x + threadIdx.x;
  if (i < n) xh[i] = (_Float16)x[i];
}

// agg = x  (f16 payload copied as dwords)
__global__ void gin_copy_u32(unsigned* __restrict__ dst, const unsigned* __restrict__ src, long n) {
  long i = (long)blockIdx.x * blockDim.x + threadIdx.x;
  if (i < n) dst[i] = src[i];
}

// ---------------------------------------------------------------------------
// agg[dst] += x[src] per edge: packed f16 atomics (global_atomic_pk_add_f16).
// One thread per (edge, half2-chunk): halves both gather bytes and atomic count
// vs an f32 pipeline. All traffic is L2-resident (x is only 6.4MB in f16).
// ---------------------------------------------------------------------------
__global__ void gin_scatter_h(const __half2* __restrict__ x2, const int* __restrict__ srcI,
                              const int* __restrict__ dstI, __half2* __restrict__ agg2, int E) {
  long i = (long)blockIdx.x * blockDim.x + threadIdx.x;
  if (i >= (long)E * 32) return;
  int e = (int)(i >> 5);
  int c = (int)(i & 31);
  atomicAdd(&agg2[(long)dstI[e] * 32 + c], x2[(long)srcI[e] * 32 + c]);
}

// ---- WMMA fragment layouts (ISA 7.12.2, wave32) ----------------------------
// A 16x32 f16: M=lane&15; slot i -> K = 16*(i>>3) + 8*(lane>>4) + (i&7)
// B 32x16 f16: N=lane&15; slot i -> K = 16*(i>>3) + 8*(lane>>4) + (i&7)
// C/D 16x16 f32: VGPR r -> M = r + 8*(lane>>4), N = lane&15

// Load an A-fragment: 16 halves live at p[0..7] and p[16..23] (two b128s).
__device__ __forceinline__ v16h load_afrag(const _Float16* p) {
  v8h lo = *(const v8h*)p;
  v8h hi = *(const v8h*)(p + 16);
  v16h a;
#pragma unroll
  for (int j = 0; j < 8; ++j) { a[j] = lo[j]; a[j + 8] = hi[j]; }
  return a;
}

// Swizzled B operand in LDS: fragment (n,kc) for lane l occupies 32 contiguous
// bytes at [((n*KC+kc)*32 + l)*16], so a B-fragment load is 2x ds_load_b128.
__device__ __forceinline__ v16h load_bfrag(const _Float16* base, int nkc, int lane) {
  return *(const v16h*)(base + ((nkc * 32) + lane) * 16);
}

// Fill the swizzled B operand from a row-major f32 weight [K = KC*32][SRCCOLS],
// zero-padding columns up to LDPAD. 8 halves per ds_store_b128.
template <int KC, int LDPAD, int SRCCOLS>
__device__ __forceinline__ void fill_bfrag(_Float16* dst, const float* __restrict__ W, int tid) {
  const int TOT = KC * 32 * LDPAD;
  for (int g = tid; g < TOT / 8; g += 128) {
    int d0 = g * 8;
    int i0 = d0 & 15;            // 0 or 8
    int ln = (d0 >> 4) & 31;
    int t  = d0 >> 9;            // = n*KC + kc
    int kc = t % KC;
    int n  = t / KC;
    int col = n * 16 + (ln & 15);
    int kb  = kc * 32 + ((ln >> 4) & 1) * 8 + i0 * 2;
    v8h v;
#pragma unroll
    for (int j = 0; j < 8; ++j)
      v[j] = (col < SRCCOLS) ? (_Float16)W[(kb + j) * SRCCOLS + col] : (_Float16)0.0f;
    *(v8h*)(dst + d0) = v;
  }
}

// 16 x 64-half contiguous tile (2KB): global -> LDS, async if available.
// OOB tail rows are skipped (garbage in LDS is harmless: matmul rows are
// independent and all stores are row-guarded).
__device__ __forceinline__ void load_tile(_Float16* ldsDst, const _Float16* gSrc,
                                          int row0, int N, int lane) {
  const char* gb = (const char*)gSrc;
  char* lb = (char*)ldsDst;
#pragma unroll
  for (int j = 0; j < 4; ++j) {
    int c = j * 32 + lane;          // 16B chunk id; 8 chunks per row
    if (row0 + (c >> 3) < N) {
#if defined(HAVE_ASYNC_LOAD)
      __builtin_amdgcn_global_load_async_to_lds_b128(
          (gv4i*)(gb + c * 16), (lv4i*)(lb + c * 16), 0, 0);
#else
      *(uint4*)(lb + c * 16) = *(const uint4*)(gb + c * 16);
#endif
    }
  }
}

// 16 x 64-half contiguous tile: LDS -> global, async if available.
__device__ __forceinline__ void store_tile(_Float16* gDst, const _Float16* ldsSrc,
                                           int row0, int N, int lane) {
  char* gb = (char*)gDst;
  const char* lb = (const char*)ldsSrc;
#pragma unroll
  for (int j = 0; j < 4; ++j) {
    int c = j * 32 + lane;
    if (row0 + (c >> 3) < N) {
#if defined(HAVE_ASYNC_STORE)
      __builtin_amdgcn_global_store_async_from_lds_b128(
          (gv4i*)(gb + c * 16), (lv4i*)(lb + c * 16), 0, 0);
#else
      *(uint4*)(gb + c * 16) = *(const uint4*)(lb + c * 16);
#endif
    }
  }
}

// ---------------------------------------------------------------------------
// Fused GIN MLP layer: xout = relu(bn2( relu(bn1( agg@W1 + b1 )) @ W2 + b2 ))
// One wave per 16-node tile; f16 in/out, f32 WMMA accumulation.
// ---------------------------------------------------------------------------
__global__ __launch_bounds__(128) void gin_mlp(
    const _Float16* __restrict__ aggh,
    const float* __restrict__ W1, const float* __restrict__ b1,
    const float* __restrict__ g1, const float* __restrict__ bt1,
    const float* __restrict__ m1, const float* __restrict__ v1,
    const float* __restrict__ W2, const float* __restrict__ b2,
    const float* __restrict__ gc, const float* __restrict__ bc,
    const float* __restrict__ mc, const float* __restrict__ vc,
    _Float16* __restrict__ xout, int N)
{
  __shared__ alignas(32) _Float16 sW1f[FDIM * H2];     // swizzled B operand
  __shared__ alignas(32) _Float16 sW2f[H2 * HDIM];     // swizzled B operand
  __shared__ float sSc1[H2], sSh1[H2], sSc2[HDIM], sSh2[HDIM];
  __shared__ alignas(16) _Float16 sIn[WAVES][16 * FDIM];   // input tile / out staging
  __shared__ alignas(16) _Float16 sMid[WAVES][16 * H2];

  const int tid  = threadIdx.x;
  const int lane = tid & 31;
  const int wave = tid >> 5;

  fill_bfrag<2, H2, H2>(sW1f, W1, tid);        // K=64,  N=128
  fill_bfrag<4, HDIM, HDIM>(sW2f, W2, tid);    // K=128, N=64
  if (tid < H2) {                              // fold bias+BN: y = z*sc + sh
    float s = g1[tid] * rsqrtf(v1[tid] + BN_EPS);
    sSc1[tid] = s;
    sSh1[tid] = (b1[tid] - m1[tid]) * s + bt1[tid];
  }
  if (tid < HDIM) {
    float s = gc[tid] * rsqrtf(vc[tid] + BN_EPS);
    sSc2[tid] = s;
    sSh2[tid] = (b2[tid] - mc[tid]) * s + bc[tid];
  }

  const int row0 = (blockIdx.x * WAVES + wave) * 16;
  load_tile(sIn[wave], aggh + (size_t)row0 * FDIM, row0, N, lane);
  wait_async0();
  __syncthreads();

  const int m   = lane & 15;
  const int kbA = ((lane >> 4) & 1) << 3;   // A k-offset AND D row offset
  const _Float16* inrow = &sIn[wave][m * FDIM + kbA];

  // ---- GEMM1: [16x64] @ [64x128], K in two 32-chunks ----
  v16h a0 = load_afrag(inrow);
  v16h a1 = load_afrag(inrow + 32);
#pragma unroll
  for (int n = 0; n < 8; ++n) {
    v16h bf0 = load_bfrag(sW1f, n * 2 + 0, lane);
    v16h bf1 = load_bfrag(sW1f, n * 2 + 1, lane);
    v8f c = {};
    c = __builtin_amdgcn_wmma_f32_16x16x32_f16(false, a0, false, bf0, (short)0, c, false, false);
    c = __builtin_amdgcn_wmma_f32_16x16x32_f16(false, a1, false, bf1, (short)0, c, false, false);
    int ncol = n * 16 + m;
    float sc = sSc1[ncol], sh = sSh1[ncol];
#pragma unroll
    for (int r = 0; r < 8; ++r) {
      float vv = c[r] * sc + sh;
      vv = vv > 0.0f ? vv : 0.0f;
      sMid[wave][(r + kbA) * H2 + ncol] = (_Float16)vv;   // same-wave DS: in order
    }
  }

  // ---- GEMM2: [16x128] @ [128x64], K in four 32-chunks ----
  v16h a2[4];
#pragma unroll
  for (int k = 0; k < 4; ++k)
    a2[k] = load_afrag(&sMid[wave][m * H2 + k * 32 + kbA]);
#pragma unroll
  for (int n = 0; n < 4; ++n) {
    v8f c = {};
#pragma unroll
    for (int k = 0; k < 4; ++k) {
      v16h bf = load_bfrag(sW2f, n * 4 + k, lane);
      c = __builtin_amdgcn_wmma_f32_16x16x32_f16(false, a2[k], false, bf, (short)0, c, false, false);
    }
    int ncol = n * 16 + m;
    float sc = sSc2[ncol], sh = sSh2[ncol];
#pragma unroll
    for (int r = 0; r < 8; ++r) {
      float vv = c[r] * sc + sh;
      vv = vv > 0.0f ? vv : 0.0f;
      sIn[wave][(r + kbA) * HDIM + ncol] = (_Float16)vv;  // reuse sIn as out staging
    }
  }

  store_tile(xout + (size_t)row0 * HDIM, sIn[wave], row0, N, lane);
  wait_async0();
}

// ---------------------------------------------------------------------------
// Head: h = relu(bn(x@lin1+b)); logits = h@lin2+b; out = log_softmax(logits)
// ---------------------------------------------------------------------------
__global__ __launch_bounds__(128) void gin_head(
    const _Float16* __restrict__ xh,
    const float* __restrict__ lin1W, const float* __restrict__ lin1b,
    const float* __restrict__ gB, const float* __restrict__ bB,
    const float* __restrict__ mB, const float* __restrict__ vB,
    const float* __restrict__ lin2W, const float* __restrict__ lin2b,
    float* __restrict__ out, int N)
{
  __shared__ alignas(32) _Float16 sW1f[HDIM * HDIM];   // swizzled, K=64 N=64
  __shared__ alignas(32) _Float16 sW2f[HDIM * CPAD];   // swizzled, K=64 N=48 (pad 0)
  __shared__ float sSc[HDIM], sSh[HDIM], sB2[CPAD];
  __shared__ alignas(16) _Float16 sIn[WAVES][16 * HDIM];
  __shared__ alignas(16) _Float16 sMid[WAVES][16 * HDIM];
  __shared__ float sLog[WAVES][16 * CPAD];

  const int tid  = threadIdx.x;
  const int lane = tid & 31;
  const int wave = tid >> 5;

  fill_bfrag<2, HDIM, HDIM>(sW1f, lin1W, tid);
  fill_bfrag<2, CPAD, CDIM>(sW2f, lin2W, tid);
  if (tid < HDIM) {
    float s = gB[tid] * rsqrtf(vB[tid] + BN_EPS);
    sSc[tid] = s;
    sSh[tid] = (lin1b[tid] - mB[tid]) * s + bB[tid];
  }
  if (tid < CPAD) sB2[tid] = (tid < CDIM) ? lin2b[tid] : 0.0f;

  const int row0 = (blockIdx.x * WAVES + wave) * 16;
  load_tile(sIn[wave], xh + (size_t)row0 * HDIM, row0, N, lane);
  wait_async0();
  __syncthreads();

  const int m   = lane & 15;
  const int kbA = ((lane >> 4) & 1) << 3;
  const _Float16* inrow = &sIn[wave][m * HDIM + kbA];

  // ---- GEMM1: [16x64]@[64x64] -> sMid (bn+relu, f16) ----
  v16h a0 = load_afrag(inrow);
  v16h a1 = load_afrag(inrow + 32);
#pragma unroll
  for (int n = 0; n < 4; ++n) {
    v16h bf0 = load_bfrag(sW1f, n * 2 + 0, lane);
    v16h bf1 = load_bfrag(sW1f, n * 2 + 1, lane);
    v8f c = {};
    c = __builtin_amdgcn_wmma_f32_16x16x32_f16(false, a0, false, bf0, (short)0, c, false, false);
    c = __builtin_amdgcn_wmma_f32_16x16x32_f16(false, a1, false, bf1, (short)0, c, false, false);
    int ncol = n * 16 + m;
    float sc = sSc[ncol], sh = sSh[ncol];
#pragma unroll
    for (int r = 0; r < 8; ++r) {
      float vv = c[r] * sc + sh;
      vv = vv > 0.0f ? vv : 0.0f;
      sMid[wave][(r + kbA) * HDIM + ncol] = (_Float16)vv;
    }
  }

  // ---- GEMM2: [16x64]@[64x48] -> sLog (+bias) ----
  v16h h0 = load_afrag(&sMid[wave][m * HDIM + kbA]);
  v16h h1 = load_afrag(&sMid[wave][m * HDIM + 32 + kbA]);
#pragma unroll
  for (int n = 0; n < 3; ++n) {
    v16h bf0 = load_bfrag(sW2f, n * 2 + 0, lane);
    v16h bf1 = load_bfrag(sW2f, n * 2 + 1, lane);
    v8f c = {};
    c = __builtin_amdgcn_wmma_f32_16x16x32_f16(false, h0, false, bf0, (short)0, c, false, false);
    c = __builtin_amdgcn_wmma_f32_16x16x32_f16(false, h1, false, bf1, (short)0, c, false, false);
    int ncol = n * 16 + m;
#pragma unroll
    for (int r = 0; r < 8; ++r)
      sLog[wave][(r + kbA) * CPAD + ncol] = c[r] + sB2[ncol];
  }

  // ---- per-row log-softmax over 40 classes (lanes 0..15, one row each) ----
  if (lane < 16) {
    long grow = (long)row0 + lane;
    if (grow < (long)N) {
      const float* rowp = &sLog[wave][lane * CPAD];
      float mx = -1e30f;
      for (int cc = 0; cc < CDIM; ++cc) mx = fmaxf(mx, rowp[cc]);
      float sum = 0.0f;
      for (int cc = 0; cc < CDIM; ++cc) sum += expf(rowp[cc] - mx);
      float lse = mx + logf(sum);
      for (int cc = 0; cc < CDIM; ++cc) out[grow * CDIM + cc] = rowp[cc] - lse;
    }
  }
}

// ---------------------------------------------------------------------------
extern "C" void kernel_launch(void* const* d_in, const int* in_sizes, int n_in,
                              void* d_out, int out_size, void* d_ws, size_t ws_size,
                              hipStream_t stream) {
  const float* x     = (const float*)d_in[0];
  const int*   ei    = (const int*)d_in[1];   // [2,E]
  const float* W1s   = (const float*)d_in[2];
  const float* b1s   = (const float*)d_in[3];
  const float* g1s   = (const float*)d_in[4];
  const float* bt1s  = (const float*)d_in[5];
  const float* m1s   = (const float*)d_in[6];
  const float* v1s   = (const float*)d_in[7];
  const float* W2s   = (const float*)d_in[8];
  const float* b2s   = (const float*)d_in[9];
  const float* gcs   = (const float*)d_in[10];
  const float* bcs   = (const float*)d_in[11];
  const float* mcs   = (const float*)d_in[12];
  const float* vcs   = (const float*)d_in[13];
  const float* lin1W = (const float*)d_in[14];
  const float* lin1b = (const float*)d_in[15];
  const float* gb1   = (const float*)d_in[16];
  const float* bb1   = (const float*)d_in[17];
  const float* mb1   = (const float*)d_in[18];
  const float* vb1   = (const float*)d_in[19];
  const float* lin2W = (const float*)d_in[20];
  const float* lin2b = (const float*)d_in[21];

  const int N = in_sizes[0] / FDIM;
  const int E = in_sizes[1] / 2;

  _Float16* xh   = (_Float16*)d_ws;                 // [N,64] f16 current features
  _Float16* aggh = xh + (size_t)N * FDIM;           // [N,64] f16 aggregation

  const long nElem   = (long)N * FDIM;
  const int  cvtBlk  = (int)((nElem + 255) / 256);
  const int  cpyBlk  = (int)((nElem / 2 + 255) / 256);
  const long eTh     = (long)E * 32;
  const int  scatBlk = (int)((eTh + 255) / 256);
  const int  tiles   = (N + 15) / 16;
  const int  gemmBlk = (tiles + WAVES - 1) / WAVES;

  gin_cvt_h<<<cvtBlk, 256, 0, stream>>>(x, xh, nElem);

  for (int l = 0; l < 3; ++l) {
    gin_copy_u32<<<cpyBlk, 256, 0, stream>>>((unsigned*)aggh, (const unsigned*)xh, nElem / 2);
    gin_scatter_h<<<scatBlk, 256, 0, stream>>>((const __half2*)xh, ei, ei + E,
                                               (__half2*)aggh, E);
    gin_mlp<<<gemmBlk, 128, 0, stream>>>(aggh,
        W1s + (size_t)l * FDIM * H2, b1s + (size_t)l * H2,
        g1s + (size_t)l * H2, bt1s + (size_t)l * H2,
        m1s + (size_t)l * H2, v1s + (size_t)l * H2,
        W2s + (size_t)l * H2 * HDIM, b2s + (size_t)l * HDIM,
        gcs + (size_t)l * HDIM, bcs + (size_t)l * HDIM,
        mcs + (size_t)l * HDIM, vcs + (size_t)l * HDIM,
        xh, N);
  }
  gin_head<<<gemmBlk, 128, 0, stream>>>(xh, lin1W, lin1b, gb1, bb1, mb1, vb1,
                                        lin2W, lin2b, (float*)d_out, N);
}
